// PriorStateTracker_20212116095186
// MI455X (gfx1250) — matrix operand order repaired and checked by reading, not compiled, not run on hardware
//
#include <hip/hip_runtime.h>
#include <math.h>

typedef __bf16 bf16;
typedef __attribute__((ext_vector_type(16))) __bf16 v16bf;
typedef __attribute__((ext_vector_type(8)))  __bf16 v8bf;
typedef __attribute__((ext_vector_type(8)))  float   v8f;
typedef __attribute__((ext_vector_type(2)))  float   v2f;

#define B_  64
#define S_  10
#define K_  30000
#define KP  30016      // K padded to multiple of 32
#define E_  512
#define H_  512
#define T_  512
#define E3  1536
#define KSPLIT 16      // K-dim parallel splits for GEMM1
#define NSPLIT 4       // column-chunk splits for attention scores
#define NEG_INF (-1000000000.0f)

// ---------------------------------------------------------------------------
// 1) pv_state (640 x 30000 f32) -> A_bf (640 x 30016 bf16, zero padded)
// ---------------------------------------------------------------------------
__global__ void k_cvt_pad_bf16(const float* __restrict__ in, bf16* __restrict__ out,
                               int rows, int incols, int outcols) {
  long long idx = (long long)blockIdx.x * blockDim.x + threadIdx.x;
  long long total = (long long)rows * outcols;
  if (idx >= total) return;
  int r = (int)(idx / outcols), c = (int)(idx % outcols);
  float v = (c < incols) ? in[(long long)r * incols + c] : 0.0f;
  out[idx] = (bf16)v;
}

// ---------------------------------------------------------------------------
// 2) gather rows embed[know2word[k]] and store TRANSPOSED bf16: Bt[e][k]
// ---------------------------------------------------------------------------
__global__ void k_gather_transpose_bf16(const float* __restrict__ embed,
                                        const int* __restrict__ k2w,
                                        bf16* __restrict__ out /* [E_][KP] */) {
  __shared__ bf16 tile[64][65];
  int k0 = blockIdx.x * 64;
  int c0 = blockIdx.y * 64;
  for (int t = threadIdx.x; t < 64 * 64; t += 256) {
    int i = t >> 6, c = t & 63;
    int k = k0 + i;
    float v = 0.0f;
    if (k < K_) { int w = k2w[k]; v = embed[(long long)w * E_ + c0 + c]; }
    tile[c][i] = (bf16)v;
  }
  __syncthreads();
  for (int t = threadIdx.x; t < 64 * 64; t += 256) {
    int c = t >> 6, i = t & 63;
    out[(long long)(c0 + c) * KP + k0 + i] = tile[c][i];
  }
}

// ---------------------------------------------------------------------------
// 3) GEMM1, K-split: Cpart[split] = A_bf(640xKP) * Bt_bf (16x64 per wave)
//    manual ping-pong (2x unroll, two named register sets -> no rotate-copies)
// ---------------------------------------------------------------------------
#define LOAD_FRAGS(AF, F0, F1, F2, F3, KK)                          \
  do {                                                              \
    ((v8bf*)&(AF))[0] = *(const v8bf*)(aRow + (KK));                \
    ((v8bf*)&(AF))[1] = *(const v8bf*)(aRow + (KK) + 8);            \
    ((v8bf*)&(F0))[0] = *(const v8bf*)(b0 + (KK));                  \
    ((v8bf*)&(F0))[1] = *(const v8bf*)(b0 + (KK) + 8);              \
    ((v8bf*)&(F1))[0] = *(const v8bf*)(b1 + (KK));                  \
    ((v8bf*)&(F1))[1] = *(const v8bf*)(b1 + (KK) + 8);              \
    ((v8bf*)&(F2))[0] = *(const v8bf*)(b2 + (KK));                  \
    ((v8bf*)&(F2))[1] = *(const v8bf*)(b2 + (KK) + 8);              \
    ((v8bf*)&(F3))[0] = *(const v8bf*)(b3 + (KK));                  \
    ((v8bf*)&(F3))[1] = *(const v8bf*)(b3 + (KK) + 8);              \
  } while (0)

#define WMMA4(AF, F0, F1, F2, F3)                                                                     \
  do {                                                                                                \
    acc0 = __builtin_amdgcn_wmma_f32_16x16x32_bf16(false, AF, false, F0, (short)0, acc0, false, false); \
    acc1 = __builtin_amdgcn_wmma_f32_16x16x32_bf16(false, AF, false, F1, (short)0, acc1, false, false); \
    acc2 = __builtin_amdgcn_wmma_f32_16x16x32_bf16(false, AF, false, F2, (short)0, acc2, false, false); \
    acc3 = __builtin_amdgcn_wmma_f32_16x16x32_bf16(false, AF, false, F3, (short)0, acc3, false, false); \
  } while (0)

__global__ void k_gemm1_bf16(const bf16* __restrict__ A, const bf16* __restrict__ Bt,
                             float* __restrict__ Cpart) {
  int wave = threadIdx.x >> 5;
  int lane = threadIdx.x & 31;
  int m = lane & 15, half = lane >> 4;
  int row0 = blockIdx.x * 16;
  int colw = wave * 64;
  int split = blockIdx.y;
  const int ITERS = KP / 32;                       // 938
  int i0 = (ITERS * split) / KSPLIT;
  int i1 = (ITERS * (split + 1)) / KSPLIT;

  v8f acc0 = {}, acc1 = {}, acc2 = {}, acc3 = {};
  const bf16* aRow = A + (long long)(row0 + m) * KP + 16 * half;
  const bf16* b0 = Bt + (long long)(colw + 0 * 16 + m) * KP + 16 * half;
  const bf16* b1 = Bt + (long long)(colw + 1 * 16 + m) * KP + 16 * half;
  const bf16* b2 = Bt + (long long)(colw + 2 * 16 + m) * KP + 16 * half;
  const bf16* b3 = Bt + (long long)(colw + 3 * 16 + m) * KP + 16 * half;

  int nIter = i1 - i0;
  int nPair = nIter >> 1;
  long long kk = (long long)i0 * 32;

  v16bf afA, fA0, fA1, fA2, fA3;
  v16bf afB, fB0, fB1, fB2, fB3;
  LOAD_FRAGS(afA, fA0, fA1, fA2, fA3, kk);
  for (int p = 0; p < nPair; ++p) {
    // mid-pair load is always in range (iteration i0+2p+1 < i1 <= ITERS)
    LOAD_FRAGS(afB, fB0, fB1, fB2, fB3, kk + 32);
    __builtin_prefetch(aRow + kk + 1024, 0, 1);     // stream-ahead on A rows
    WMMA4(afA, fA0, fA1, fA2, fA3);
    long long k2 = kk + 64;
    if (k2 > (long long)KP - 32) k2 = (long long)KP - 32;  // wrap reload (unused data)
    LOAD_FRAGS(afA, fA0, fA1, fA2, fA3, k2);
    WMMA4(afB, fB0, fB1, fB2, fB3);
    kk += 64;
  }
  if (nIter & 1) {
    WMMA4(afA, fA0, fA1, fA2, fA3);
  }

  float* Cp = Cpart + (long long)split * (B_ * S_ * E_);
#pragma unroll
  for (int r = 0; r < 8; ++r) {
    int row = row0 + r + 8 * half;
    Cp[(long long)row * E_ + colw + 0 * 16 + m] = acc0[r];
    Cp[(long long)row * E_ + colw + 1 * 16 + m] = acc1[r];
    Cp[(long long)row * E_ + colw + 2 * 16 + m] = acc2[r];
    Cp[(long long)row * E_ + colw + 3 * 16 + m] = acc3[r];
  }
}

// fixed-order (deterministic) reduction of the K-split partials
__global__ void k_reduce_c1(const float* __restrict__ Cpart, float* __restrict__ C1) {
  int i = blockIdx.x * blockDim.x + threadIdx.x;   // 0 .. 640*512
  float s = 0.0f;
#pragma unroll
  for (int p = 0; p < KSPLIT; ++p) s += Cpart[(long long)p * (B_ * S_ * E_) + i];
  C1[i] = s;
}

// ---------------------------------------------------------------------------
// Generic f32 WMMA GEMM (16x16x4): C = ascale*A @ B (+ bias)
// two interleaved accumulators to break the WMMA dependency chain
// ---------------------------------------------------------------------------
__global__ void k_gemm_f32w(const float* __restrict__ A, const float* __restrict__ Bm,
                            const float* __restrict__ bias, float* __restrict__ C,
                            int M, int K, int N, float ascale) {
  int gw = (int)((blockIdx.x * blockDim.x + threadIdx.x) >> 5);
  int lane = threadIdx.x & 31;
  int ntiles = N >> 4;
  int mt = gw / ntiles, nt = gw % ntiles;
  if (mt >= (M >> 4)) return;
  int m = lane & 15, half = lane >> 4;
  int row = mt * 16 + m;
  int col = nt * 16 + m;
  v8f accA = {}, accB = {};
  const float* ap = A + (long long)row * K + 2 * half;
  const float* bp = Bm + (long long)(2 * half) * N + col;
  for (int k0 = 0; k0 < K; k0 += 8) {
    v2f a0, b0v, a1, b1v;
    a0[0] = ap[k0] * ascale;
    a0[1] = ap[k0 + 1] * ascale;
    b0v[0] = bp[(long long)k0 * N];
    b0v[1] = bp[(long long)(k0 + 1) * N];
    a1[0] = ap[k0 + 4] * ascale;
    a1[1] = ap[k0 + 5] * ascale;
    b1v[0] = bp[(long long)(k0 + 4) * N];
    b1v[1] = bp[(long long)(k0 + 5) * N];
    accA = __builtin_amdgcn_wmma_f32_16x16x4_f32(false, a0, false, b0v, (short)0, accA, false, false);
    accB = __builtin_amdgcn_wmma_f32_16x16x4_f32(false, a1, false, b1v, (short)0, accB, false, false);
  }
  v8f acc = accA + accB;
  float bv = bias ? bias[col] : 0.0f;
#pragma unroll
  for (int r = 0; r < 8; ++r)
    C[(long long)(mt * 16 + r + 8 * half) * N + col] = acc[r] + bv;
}

// ---------------------------------------------------------------------------
// f32 transpose helpers
// ---------------------------------------------------------------------------
__global__ void k_transpose_f32(const float* __restrict__ in, float* __restrict__ out,
                                int R, int Cc) {
  __shared__ float tile[32][33];
  int r0 = blockIdx.y * 32, c0 = blockIdx.x * 32;
  int tx = threadIdx.x & 31, ty = threadIdx.x >> 5;
  for (int i = ty; i < 32; i += 8) {
    int r = r0 + i, c = c0 + tx;
    tile[i][tx] = (r < R && c < Cc) ? in[(long long)r * Cc + c] : 0.0f;
  }
  __syncthreads();
  for (int i = ty; i < 32; i += 8) {
    int c = c0 + i, r = r0 + tx;
    if (c < Cc && r < R) out[(long long)c * R + r] = tile[tx][i];
  }
}

__global__ void k_transpose_f32_bf16(const float* __restrict__ in, bf16* __restrict__ out,
                                     int R, int Cc) {
  __shared__ float tile[32][33];
  int r0 = blockIdx.y * 32, c0 = blockIdx.x * 32;
  int tx = threadIdx.x & 31, ty = threadIdx.x >> 5;
  for (int i = ty; i < 32; i += 8) {
    int r = r0 + i, c = c0 + tx;
    tile[i][tx] = (r < R && c < Cc) ? in[(long long)r * Cc + c] : 0.0f;
  }
  __syncthreads();
  for (int i = ty; i < 32; i += 8) {
    int c = c0 + i, r = r0 + tx;
    if (c < Cc && r < R) out[(long long)c * R + r] = (bf16)tile[tx][i];
  }
}

__global__ void k_cvt_bf16(const float* __restrict__ in, bf16* __restrict__ out, long long n) {
  long long i = (long long)blockIdx.x * blockDim.x + threadIdx.x;
  if (i < n) out[i] = (bf16)in[i];
}

__global__ void k_zero(float* __restrict__ p, int n) {
  int i = blockIdx.x * blockDim.x + threadIdx.x;
  if (i < n) p[i] = 0.0f;
}

// ---------------------------------------------------------------------------
// GRU gate combine (per step); accumulates q = sum_s h_s
// ---------------------------------------------------------------------------
__global__ void k_gru_combine(const float* __restrict__ gi, const float* __restrict__ gh,
                              float* __restrict__ h, float* __restrict__ qacc, int s) {
  int idx = blockIdx.x * blockDim.x + threadIdx.x;   // 0 .. 64*512
  int b = idx >> 9, e = idx & 511;
  const float* gib = gi + (long long)(b * S_ + s) * E3;
  const float* ghb = gh + (long long)b * E3;
  float ir = gib[e], iz = gib[E_ + e], in_ = gib[2 * E_ + e];
  float hr = ghb[e], hz = ghb[E_ + e], hn = ghb[2 * E_ + e];
  float r = 1.0f / (1.0f + expf(-(ir + hr)));
  float z = 1.0f / (1.0f + expf(-(iz + hz)));
  float n = tanhf(in_ + r * hn);
  float hnew = (1.0f - z) * n + z * h[idx];
  h[idx] = hnew;
  qacc[idx] = (s == 0 ? 0.0f : qacc[idx]) + hnew;
}

// ---------------------------------------------------------------------------
// Fused attention scores, split over NSPLIT column chunks.
// Ping-pong double buffering + dual accumulators inside the K loop.
// tanh applied after the FULL K=512 reduction per column tile -> exact.
// ---------------------------------------------------------------------------
__global__ void k_attn_scores(const bf16* __restrict__ encb, const bf16* __restrict__ WkT,
                              const float* __restrict__ qWq, const float* __restrict__ b_attn,
                              const float* __restrict__ v_attn, float* __restrict__ spart) {
  int gw = (int)((blockIdx.x * blockDim.x + threadIdx.x) >> 5);  // 0..8191
  int lane = threadIdx.x & 31;
  int m = lane & 15, half = lane >> 4;
  int tileId = gw >> 2;        // bt tile 0..2047
  int part = gw & 3;           // column chunk 0..3
  int bt0 = tileId * 16;
  int b = bt0 >> 9;            // T = 512, tiles never cross batch rows
  const bf16* aRow = encb + (long long)(bt0 + m) * H_ + 16 * half;
  float sacc[8] = {0, 0, 0, 0, 0, 0, 0, 0};
  int nt0 = part * (H_ / 16 / NSPLIT);  // 8 tiles per chunk
  for (int nt = nt0; nt < nt0 + H_ / 16 / NSPLIT; ++nt) {
    int col = nt * 16 + m;
    const bf16* bp = WkT + (long long)col * H_ + 16 * half;
    v8f accA = {}, accB = {};
    v16bf aA, bA, aB, bB;
    ((v8bf*)&aA)[0] = *(const v8bf*)(aRow);
    ((v8bf*)&aA)[1] = *(const v8bf*)(aRow + 8);
    ((v8bf*)&bA)[0] = *(const v8bf*)(bp);
    ((v8bf*)&bA)[1] = *(const v8bf*)(bp + 8);
    for (int k0 = 0; k0 < H_; k0 += 64) {
      ((v8bf*)&aB)[0] = *(const v8bf*)(aRow + k0 + 32);
      ((v8bf*)&aB)[1] = *(const v8bf*)(aRow + k0 + 40);
      ((v8bf*)&bB)[0] = *(const v8bf*)(bp + k0 + 32);
      ((v8bf*)&bB)[1] = *(const v8bf*)(bp + k0 + 40);
      accA = __builtin_amdgcn_wmma_f32_16x16x32_bf16(false, aA, false, bA, (short)0, accA, false, false);
      int k2 = k0 + 64;
      if (k2 > H_ - 32) k2 = H_ - 32;          // wrap reload (data unused)
      ((v8bf*)&aA)[0] = *(const v8bf*)(aRow + k2);
      ((v8bf*)&aA)[1] = *(const v8bf*)(aRow + k2 + 8);
      ((v8bf*)&bA)[0] = *(const v8bf*)(bp + k2);
      ((v8bf*)&bA)[1] = *(const v8bf*)(bp + k2 + 8);
      accB = __builtin_amdgcn_wmma_f32_16x16x32_bf16(false, aB, false, bB, (short)0, accB, false, false);
    }
    float add = qWq[(long long)b * H_ + col] + b_attn[col];
    float va = v_attn[col];
#pragma unroll
    for (int r = 0; r < 8; ++r)
      sacc[r] += tanhf(accA[r] + accB[r] + add) * va;
  }
#pragma unroll
  for (int r = 0; r < 8; ++r) {
    float v = sacc[r];
    v += __shfl_xor(v, 1);
    v += __shfl_xor(v, 2);
    v += __shfl_xor(v, 4);
    v += __shfl_xor(v, 8);
    sacc[r] = v;
  }
  if (m == 0) {
#pragma unroll
    for (int r = 0; r < 8; ++r)
      spart[(long long)part * (B_ * T_) + bt0 + 8 * half + r] = sacc[r];
  }
}

// ---------------------------------------------------------------------------
// masked softmax over T per batch row (sums the NSPLIT score partials)
// ---------------------------------------------------------------------------
__global__ void k_softmax(const float* __restrict__ spart, const int* __restrict__ pv_r_u,
                          float* __restrict__ w) {
  __shared__ float red[256];
  int b = blockIdx.x;
  int tid = threadIdx.x;
  float s0 = 0.0f, s1 = 0.0f;
#pragma unroll
  for (int p = 0; p < NSPLIT; ++p) {
    s0 += spart[(long long)p * (B_ * T_) + b * T_ + tid];
    s1 += spart[(long long)p * (B_ * T_) + b * T_ + 256 + tid];
  }
  if (pv_r_u[b * T_ + tid] <= 4) s0 = NEG_INF;
  if (pv_r_u[b * T_ + 256 + tid] <= 4) s1 = NEG_INF;
  red[tid] = fmaxf(s0, s1);
  __syncthreads();
  for (int st = 128; st > 0; st >>= 1) {
    if (tid < st) red[tid] = fmaxf(red[tid], red[tid + st]);
    __syncthreads();
  }
  float mx = red[0];
  __syncthreads();
  float e0 = expf(s0 - mx), e1 = expf(s1 - mx);
  red[tid] = e0 + e1;
  __syncthreads();
  for (int st = 128; st > 0; st >>= 1) {
    if (tid < st) red[tid] += red[tid + st];
    __syncthreads();
  }
  float inv = 1.0f / red[0];
  w[b * T_ + tid] = e0 * inv;
  w[b * T_ + 256 + tid] = e1 * inv;
}

// ---------------------------------------------------------------------------
// fused[b,h] = hidden[b,h] + sum_t w[b,t] * enc[b,t,h]
// ---------------------------------------------------------------------------
__global__ void k_fused(const float* __restrict__ hidden, const float* __restrict__ enc,
                        const float* __restrict__ w, float* __restrict__ out) {
  __shared__ float wl[T_];
  int b = blockIdx.x, chunk = blockIdx.y;
  int tid = threadIdx.x;  // 128
  for (int t = tid; t < T_; t += 128) wl[t] = w[b * T_ + t];
  __syncthreads();
  int hcol = chunk * 128 + tid;
  float acc = hidden[b * H_ + hcol];
  const float* eb = enc + (long long)b * T_ * H_ + hcol;
  for (int t = 0; t < T_; ++t) acc += wl[t] * eb[(long long)t * H_];
  out[b * H_ + hcol] = acc;
}

// ---------------------------------------------------------------------------
extern "C" void kernel_launch(void* const* d_in, const int* in_sizes, int n_in,
                              void* d_out, int out_size, void* d_ws, size_t ws_size,
                              hipStream_t stream) {
  const float* pv_state  = (const float*)d_in[0];
  const float* hidden    = (const float*)d_in[1];
  const float* enc       = (const float*)d_in[2];
  const float* embed     = (const float*)d_in[3];
  const float* W_ih      = (const float*)d_in[4];
  const float* W_hh      = (const float*)d_in[5];
  const float* b_ih      = (const float*)d_in[6];
  const float* b_hh      = (const float*)d_in[7];
  const float* Wq        = (const float*)d_in[8];
  const float* Wk        = (const float*)d_in[9];
  const float* b_attn    = (const float*)d_in[10];
  const float* v_attn    = (const float*)d_in[11];
  const int*   pv_r_u    = (const int*)d_in[12];
  const int*   know2word = (const int*)d_in[13];
  float* out = (float*)d_out;

  char* base = (char*)d_ws;
  size_t off = 0;
  auto take = [&](size_t bytes) { size_t r = off; off += (bytes + 255) & ~(size_t)255; return r; };
  size_t oA  = take((size_t)(B_ * S_) * KP * sizeof(bf16));   // 38.4 MB  (later: enc_bf16)
  size_t oBt = take((size_t)E_ * KP * sizeof(bf16));          // 30.7 MB  (later: WkT_bf16)
  size_t oC1 = take((size_t)(B_ * S_) * E_ * sizeof(float));
  // --- union region: Cpart (dead after reduce) overlaps all later small bufs
  size_t oU = off;
  size_t oCp = oU;
  size_t endCp = oU + (((size_t)KSPLIT * B_ * S_ * E_ * sizeof(float) + 255) & ~(size_t)255);
  off = oU;
  size_t oWih = take((size_t)E_ * E3 * sizeof(float));
  size_t oWhh = take((size_t)E_ * E3 * sizeof(float));
  size_t oGi  = take((size_t)(B_ * S_) * E3 * sizeof(float));
  size_t oGh  = take((size_t)B_ * E3 * sizeof(float));
  size_t oH   = take((size_t)B_ * E_ * sizeof(float));
  size_t oQa  = take((size_t)B_ * E_ * sizeof(float));
  size_t oQW  = take((size_t)B_ * H_ * sizeof(float));
  size_t oSp  = take((size_t)NSPLIT * B_ * T_ * sizeof(float));
  size_t oW   = take((size_t)B_ * T_ * sizeof(float));
  if (off < endCp) off = endCp;
  size_t oEnc = oA;    // alias: enc_bf16 (33.5 MB <= 38.4 MB), used after GEMM1
  size_t oWkT = oBt;   // alias: WkT_bf16, used after GEMM1

  bf16*  A_bf   = (bf16*)(base + oA);
  bf16*  Bt_bf  = (bf16*)(base + oBt);
  float* C1     = (float*)(base + oC1);
  float* Cpart  = (float*)(base + oCp);
  float* WihT   = (float*)(base + oWih);
  float* WhhT   = (float*)(base + oWhh);
  float* gi     = (float*)(base + oGi);
  float* gh     = (float*)(base + oGh);
  float* hbuf   = (float*)(base + oH);
  float* qacc   = (float*)(base + oQa);
  float* qWqb   = (float*)(base + oQW);
  float* spart  = (float*)(base + oSp);
  float* wsm    = (float*)(base + oW);
  bf16*  enc_bf = (bf16*)(base + oEnc);
  bf16*  WkT_bf = (bf16*)(base + oWkT);

  // 1) stage bf16 operands of the big GEMM
  {
    long long tot = (long long)(B_ * S_) * KP;
    k_cvt_pad_bf16<<<(int)((tot + 255) / 256), 256, 0, stream>>>(pv_state, A_bf, B_ * S_, K_, KP);
    k_gather_transpose_bf16<<<dim3(KP / 64, E_ / 64), 256, 0, stream>>>(embed, know2word, Bt_bf);
  }
  // 2) GEMM1 (K-split x16, 5120 waves) + deterministic reduce
  k_gemm1_bf16<<<dim3((B_ * S_) / 16, KSPLIT), 256, 0, stream>>>(A_bf, Bt_bf, Cpart);
  k_reduce_c1<<<(B_ * S_ * E_) / 256, 256, 0, stream>>>(Cpart, C1);

  // 3) gi = pv_state_emb @ W_ih^T + b_ih  (f32 WMMA)
  k_transpose_f32<<<dim3(E_ / 32, E3 / 32), 256, 0, stream>>>(W_ih, WihT, E3, E_);
  k_transpose_f32<<<dim3(E_ / 32, E3 / 32), 256, 0, stream>>>(W_hh, WhhT, E3, E_);
  {
    int waves = ((B_ * S_) / 16) * (E3 / 16);  // 3840
    k_gemm_f32w<<<waves * 32 / 256, 256, 0, stream>>>(C1, WihT, b_ih, gi, B_ * S_, E_, E3, 1.0f);
  }

  // 4) GRU recurrence, 10 steps (f32 WMMA for h @ W_hh^T)
  k_zero<<<(B_ * E_) / 256, 256, 0, stream>>>(hbuf, B_ * E_);
  for (int s = 0; s < S_; ++s) {
    int waves = (B_ / 16) * (E3 / 16);  // 384
    k_gemm_f32w<<<waves * 32 / 256, 256, 0, stream>>>(hbuf, WhhT, b_hh, gh, B_, E_, E3, 1.0f);
    k_gru_combine<<<(B_ * E_) / 256, 256, 0, stream>>>(gi, gh, hbuf, qacc, s);
  }

  // 5) qWq = (qacc/S) @ Wq  (f32 WMMA; Wq already K-major)
  {
    int waves = (B_ / 16) * (H_ / 16);  // 128
    k_gemm_f32w<<<waves * 32 / 256, 256, 0, stream>>>(qacc, Wq, nullptr, qWqb, B_, E_, H_, 1.0f / (float)S_);
  }

  // 6) attention: bf16 staging (aliases GEMM1 regions) + fused score GEMM
  {
    long long n = (long long)B_ * T_ * H_;
    k_cvt_bf16<<<(int)((n + 255) / 256), 256, 0, stream>>>(enc, enc_bf, n);
    k_transpose_f32_bf16<<<dim3(H_ / 32, H_ / 32), 256, 0, stream>>>(Wk, WkT_bf, H_, H_);
    int waves = (B_ * T_) / 16 * NSPLIT;  // 8192
    k_attn_scores<<<waves * 32 / 256, 256, 0, stream>>>(enc_bf, WkT_bf, qWqb, b_attn, v_attn, spart);
  }

  // 7) masked softmax + weighted sum + residual
  k_softmax<<<B_, 256, 0, stream>>>(spart, pv_r_u, wsm);
  k_fused<<<dim3(B_, H_ / 128), 128, 0, stream>>>(hidden, enc, wsm, out);

  (void)in_sizes; (void)n_in; (void)out_size; (void)ws_size;
}
